// MultiHeadAttention_7971459301806
// MI455X (gfx1250) — matrix operand order repaired and checked by reading, not compiled, not run on hardware
//
#include <hip/hip_runtime.h>
#include <hip/hip_bf16.h>

// ---------------------------------------------------------------------------
// Types for CDNA5 WMMA (wave32)
// ---------------------------------------------------------------------------
typedef __bf16 bf16;
typedef __attribute__((ext_vector_type(16))) __bf16        v16bf;
typedef __attribute__((ext_vector_type(8)))  __bf16        bf16x8;
typedef __attribute__((ext_vector_type(8)))  float         v8f;
typedef __attribute__((ext_vector_type(4)))  float         v4f;
typedef __attribute__((ext_vector_type(16))) unsigned short u16x16;

#define SEQ   1024
#define DM    1024
#define HEADS 16
#define DK    64
#define BATCH 4
#define MROWS (BATCH * SEQ)   // 4096

// Padded LDS row for the staged 64-wide A tile: 64 data bf16 + 8 pad = 144B
// row stride. Keeps 16B alignment for b128 ops and de-phases LDS banks
// (word stride 36 -> all 16 fragment rows start in distinct banks).
#define A_ROW 72

__device__ __forceinline__ unsigned short f32_to_bf16_bits(float x) {
  unsigned int u = __float_as_uint(x);
  u += 0x7FFFu + ((u >> 16) & 1u);   // round-to-nearest-even
  return (unsigned short)(u >> 16);
}
__device__ __forceinline__ bf16 f32_to_bf16(float x) {
  union { unsigned short u; bf16 h; } cv;
  cv.u = f32_to_bf16_bits(x);
  return cv.h;
}

__device__ __forceinline__ v8f wmma_bf16(v16bf a, v16bf b, v8f c) {
  // v_wmma_f32_16x16x32_bf16: D = A(16x32) * B(32x16) + C(16x16 f32)
  return __builtin_amdgcn_wmma_f32_16x16x32_bf16(false, a, false, b,
                                                 (short)0, c, false, false);
}

// A-matrix fragment (16x32 bf16): each lane holds two contiguous 8-element
// chunks 16 elements apart in K (per ISA 7.12.2 layout). Works for global or
// LDS source pointers.
__device__ __forceinline__ v16bf load_a_frag(const bf16* p) {
  bf16x8 lo = *(const bf16x8*)p;
  bf16x8 hi = *(const bf16x8*)(p + 16);
  return __builtin_shufflevector(lo, hi, 0, 1, 2, 3, 4, 5, 6, 7,
                                 8, 9, 10, 11, 12, 13, 14, 15);
}

// Same A-fragment pattern but sourced from fp32 LDS, converted to bf16.
__device__ __forceinline__ v16bf cvt_a_frag_f32(const float* p) {
  union { u16x16 u; v16bf b; } cv;
#pragma unroll
  for (int e = 0; e < 8; ++e) {
    cv.u[e]     = f32_to_bf16_bits(p[e]);
    cv.u[e + 8] = f32_to_bf16_bits(p[16 + e]);
  }
  return cv.b;
}

// LDS byte offset of a generic pointer to __shared__ memory: the flat LDS
// aperture carries the wave-relative LDS offset in the low 32 bits.
__device__ __forceinline__ unsigned lds_offset(const void* p) {
  return (unsigned)(size_t)p;
}

// Async DMA of one 16B chunk from global memory into LDS (ASYNCcnt-tracked,
// no VGPR data return). GV addressing: 64-bit address in VGPR pair.
__device__ __forceinline__ void async_copy_b128(const void* gaddr, unsigned lds_off) {
  asm volatile("global_load_async_to_lds_b128 %0, %1, off"
               :: "v"(lds_off), "v"(gaddr) : "memory");
}
__device__ __forceinline__ void wait_async0() {
  asm volatile("s_wait_asynccnt 0x0" ::: "memory");
}

// ---------------------------------------------------------------------------
// fp32 -> bf16 bulk convert. Inputs are read exactly once -> non-temporal
// loads keep the 48MB of fp32 activations from displacing L2 working set.
// ---------------------------------------------------------------------------
__global__ void __launch_bounds__(256)
cvt_f32_bf16_kernel(const float* __restrict__ src, bf16* __restrict__ dst, int n) {
  int i = (blockIdx.x * 256 + threadIdx.x) * 4;
  if (i + 3 < n) {
    v4f f = __builtin_nontemporal_load((const v4f*)(src + i));
    uint2 u;
    u.x = (unsigned)f32_to_bf16_bits(f.x) | ((unsigned)f32_to_bf16_bits(f.y) << 16);
    u.y = (unsigned)f32_to_bf16_bits(f.z) | ((unsigned)f32_to_bf16_bits(f.w) << 16);
    *(uint2*)(dst + i) = u;
  }
}

// ---------------------------------------------------------------------------
// GEMM: C[M,N] = A[M,K] @ W[N,K]^T + bias[N]   (torch Linear semantics)
// 256 threads = 8 wave32; each wave computes a 32x32 tile (2x2 WMMA tiles);
// workgroup computes 64(M) x 128(N).
//
// The shared 64x64 A tile is DMA'd into LDS with global_load_async_to_lds_b128
// (double buffered, overlapped with WMMA): one barrier + asynccnt wait per
// 8 WMMAs per wave. B operands are loaded directly from global (L2-resident).
// ---------------------------------------------------------------------------
__global__ void __launch_bounds__(256)
gemm_nt_bf16_kernel(const bf16* __restrict__ A, const bf16* __restrict__ W,
                    const float* __restrict__ bias,
                    bf16* __restrict__ outB, float* __restrict__ outF,
                    bf16* __restrict__ outT, int M, int N, int K) {
  __shared__ __align__(16) bf16 a_sh[2 * 64 * A_ROW];   // double-buffered, 18KB

  const int n0   = blockIdx.x * 128;
  const int m0   = blockIdx.y * 64;
  const int t    = threadIdx.x;
  const int w    = t >> 5;
  const int lane = t & 31;
  const int half = lane >> 4;       // 0: lanes 0-15, 1: lanes 16-31
  const int l    = lane & 15;
  const int mloc  = (w & 1) * 32;   // wave's local m offset within the A tile
  const int nbase = n0 + (w >> 1) * 32;

  // async-copy role: thread t moves 2x16B of row (t/4) per 64-wide k-step
  const int c_row = t >> 2;            // 0..63
  const int c_chk = (t & 3) * 8;       // 0,8,16,24 (bf16 elements)
  const bf16* c_src = A + (size_t)(m0 + c_row) * K + c_chk;
  const unsigned c_dst0 = lds_offset(&a_sh[c_row * A_ROW + c_chk]);
  const unsigned c_dst1 = lds_offset(&a_sh[(64 + c_row) * A_ROW + c_chk]);

  // prologue: tile k=0 -> buffer 0 (chunks at +0 and +32 elements)
  async_copy_b128(c_src, c_dst0);
  async_copy_b128(c_src + 32, c_dst0 + 64);

  v8f acc[2][2] = {};
  for (int k = 0; k < K; k += 64) {
    const int buf = (k >> 6) & 1;
    wait_async0();                     // my async stores to LDS done
    __syncthreads();                   // whole tile visible to all waves
    if (k + 64 < K) {                  // DMA next tile into the other buffer
      const unsigned d = buf ? c_dst0 : c_dst1;
      async_copy_b128(c_src + (k + 64), d);
      async_copy_b128(c_src + (k + 96), d + 64);
    }

    const bf16* ab = &a_sh[buf * 64 * A_ROW];
#pragma unroll
    for (int kk = 0; kk < 64; kk += 32) {
      v16bf a0 = load_a_frag(ab + (mloc + l) * A_ROW + kk + half * 8);
      v16bf a1 = load_a_frag(ab + (mloc + 16 + l) * A_ROW + kk + half * 8);
      // B (32x16): lane holds one column (= one W row), two 8-element K chunks 16 apart (same as A).
      v16bf b0 = load_a_frag(W + (size_t)(nbase + l) * K + k + kk + half * 8);
      v16bf b1 = load_a_frag(W + (size_t)(nbase + 16 + l) * K + k + kk + half * 8);
      acc[0][0] = wmma_bf16(a0, b0, acc[0][0]);
      acc[0][1] = wmma_bf16(a0, b1, acc[0][1]);
      acc[1][0] = wmma_bf16(a1, b0, acc[1][0]);
      acc[1][1] = wmma_bf16(a1, b1, acc[1][1]);
    }
  }

  const int mbase = m0 + mloc;
#pragma unroll
  for (int nt = 0; nt < 2; ++nt) {
    const int col = nbase + nt * 16 + l;
    const float bv = bias ? bias[col] : 0.0f;
#pragma unroll
    for (int mt = 0; mt < 2; ++mt) {
#pragma unroll
      for (int r = 0; r < 8; ++r) {
        const int row = mbase + mt * 16 + r + half * 8;
        const float v = acc[mt][nt][r] + bv;
        if (outF)  // final result: streamed out once, never re-read -> NT
          __builtin_nontemporal_store(v, &outF[(size_t)row * N + col]);
        if (outB) outB[(size_t)row * N + col] = f32_to_bf16(v);
        if (outT) outT[(size_t)col * M + row] = f32_to_bf16(v);
      }
    }
  }
}

// ---------------------------------------------------------------------------
// Fused causal attention for one (batch, head, 16-row block):
//   scores = q @ k^T / 8  -> LDS (16 x 1024 f32, 64KB)
//   causal mask (skips fully-masked j-tiles), softmax in LDS
//   coalesced non-temporal write of probs (268MB, write-once traffic)
//   attnv = P @ v via WMMA against pre-transposed v (vt[DM][MROWS])
// ---------------------------------------------------------------------------
__global__ void __launch_bounds__(256)
attn_kernel(const bf16* __restrict__ q, const bf16* __restrict__ k_,
            const bf16* __restrict__ vt, float* __restrict__ attn_out,
            bf16* __restrict__ attnv) {
  __shared__ __align__(16) float sc[16 * SEQ];

  const int ib = blockIdx.x;            // 0..63, row block
  const int bh = blockIdx.y;            // 0..63, b*16+h
  const int b = bh >> 4, h = bh & 15;
  const int i0 = ib * 16;
  const int hbase = h * DK;
  const int w    = threadIdx.x >> 5;
  const int lane = threadIdx.x & 31;
  const int half = lane >> 4;
  const int l    = lane & 15;
  const int ntile = ib + 1;                 // causal: tiles with any valid col
  const int nt2   = (ntile + 1) & ~1;       // round up to even (K%32==0 later)
  const int wv    = nt2 * 16;               // valid score width

  // --- Phase 1: scores into LDS (each wave strides over j-tiles) ---
  v16bf aq0 = load_a_frag(q + (size_t)(b * SEQ + i0 + l) * DM + hbase + 0  + half * 8);
  v16bf aq1 = load_a_frag(q + (size_t)(b * SEQ + i0 + l) * DM + hbase + 32 + half * 8);
  for (int jt = w; jt < nt2; jt += 8) {
    const int j0 = jt * 16;
    v8f acc = {};
    v16bf kb0 = load_a_frag(k_ + (size_t)(b * SEQ + j0 + l) * DM + hbase + 0  + half * 8);
    v16bf kb1 = load_a_frag(k_ + (size_t)(b * SEQ + j0 + l) * DM + hbase + 32 + half * 8);
    acc = wmma_bf16(aq0, kb0, acc);
    acc = wmma_bf16(aq1, kb1, acc);
    const int col = j0 + l;
#pragma unroll
    for (int r = 0; r < 8; ++r) {
      const int row = r + half * 8;         // local query row 0..15
      float s = acc[r] * 0.125f;            // 1/sqrt(64)
      if (col > i0 + row) s = -1e30f;       // causal mask
      sc[row * SEQ + col] = s;
    }
  }
  __syncthreads();

  // --- Phase 2: softmax per row (wave w owns rows 2w / 2w+1) ---
  {
    const int row = 2 * w + half;
    float* sr = sc + row * SEQ;
    float mx = -1e30f;
    for (int t = 0; t < nt2; ++t) mx = fmaxf(mx, sr[l + 16 * t]);
#pragma unroll
    for (int m = 1; m < 16; m <<= 1) mx = fmaxf(mx, __shfl_xor(mx, m));
    float sum = 0.0f;
    for (int t = 0; t < nt2; ++t) {
      float e = __expf(sr[l + 16 * t] - mx);
      sr[l + 16 * t] = e;
      sum += e;
    }
#pragma unroll
    for (int m = 1; m < 16; m <<= 1) sum += __shfl_xor(sum, m);
    const float inv = 1.0f / sum;
    for (int t = 0; t < nt2; ++t) sr[l + 16 * t] *= inv;
  }
  __syncthreads();

  // --- Phase 2b: coalesced non-temporal store of probabilities ---
  {
    float* dst = attn_out + ((size_t)bh * SEQ + i0) * SEQ;
    for (int idx = threadIdx.x; idx < (16 * SEQ) / 4; idx += 256) {
      const int j  = (idx * 4) & (SEQ - 1);
      const int rr = (idx * 4) >> 10;
      v4f val = {0.f, 0.f, 0.f, 0.f};
      if (j < wv) val = *(const v4f*)&sc[rr * SEQ + j];
      __builtin_nontemporal_store(val, (v4f*)(dst + (size_t)rr * SEQ + j));
    }
  }

  // --- Phase 3: attnv(16 x 64) = P @ v ; waves 0..3 each take 16 d-columns ---
  if (w < 4) {
    const int d0 = w * 16;
    v8f acc = {};
    for (int j = 0; j < wv; j += 32) {
      v16bf pa = cvt_a_frag_f32(sc + l * SEQ + j + half * 8);
      v16bf vb = load_a_frag(vt + (size_t)(hbase + d0 + l) * MROWS +
                                 b * SEQ + j + half * 8);
      acc = wmma_bf16(pa, vb, acc);
    }
#pragma unroll
    for (int r = 0; r < 8; ++r) {
      const int row = b * SEQ + i0 + r + half * 8;
      attnv[(size_t)row * DM + hbase + d0 + l] = f32_to_bf16(acc[r]);
    }
  }
}

// ---------------------------------------------------------------------------
// Host launcher
// ---------------------------------------------------------------------------
extern "C" void kernel_launch(void* const* d_in, const int* in_sizes, int n_in,
                              void* d_out, int out_size, void* d_ws, size_t ws_size,
                              hipStream_t stream) {
  (void)in_sizes; (void)n_in; (void)out_size; (void)ws_size;
  const float* Q   = (const float*)d_in[0];
  const float* K   = (const float*)d_in[1];
  const float* V   = (const float*)d_in[2];
  /* d_in[3] = mask: causal, known statically */
  const float* W_q = (const float*)d_in[4];
  const float* b_q = (const float*)d_in[5];
  const float* W_k = (const float*)d_in[6];
  const float* b_k = (const float*)d_in[7];
  const float* W_v = (const float*)d_in[8];
  const float* b_v = (const float*)d_in[9];
  const float* W_o = (const float*)d_in[10];
  const float* b_o = (const float*)d_in[11];

  // workspace layout (all bf16, 2B each)
  char* ws = (char*)d_ws;
  const size_t ACT = (size_t)MROWS * DM * sizeof(bf16);  // 8 MB
  const size_t WGT = (size_t)DM * DM * sizeof(bf16);     // 2 MB
  bf16* Qb    = (bf16*)(ws);
  bf16* Kb    = (bf16*)(ws + ACT);
  bf16* Vb    = (bf16*)(ws + 2 * ACT);
  bf16* Wqb   = (bf16*)(ws + 3 * ACT);
  bf16* Wkb   = (bf16*)(ws + 3 * ACT + WGT);
  bf16* Wvb   = (bf16*)(ws + 3 * ACT + 2 * WGT);
  bf16* Wob   = (bf16*)(ws + 3 * ACT + 3 * WGT);
  bf16* qproj = (bf16*)(ws + 3 * ACT + 4 * WGT);
  bf16* kproj = (bf16*)(ws + 4 * ACT + 4 * WGT);
  bf16* vT    = (bf16*)(ws + 5 * ACT + 4 * WGT);   // [DM][MROWS]
  bf16* attnv = (bf16*)(ws + 6 * ACT + 4 * WGT);

  float* out_f  = (float*)d_out;                               // [4096,1024]
  float* attn_f = (float*)d_out + (size_t)MROWS * DM;          // [4,16,1024,1024]

  const int nAct = MROWS * DM;   // 4,194,304
  const int nWgt = DM * DM;      // 1,048,576
  cvt_f32_bf16_kernel<<<nAct / 1024, 256, 0, stream>>>(Q, Qb, nAct);
  cvt_f32_bf16_kernel<<<nAct / 1024, 256, 0, stream>>>(K, Kb, nAct);
  cvt_f32_bf16_kernel<<<nAct / 1024, 256, 0, stream>>>(V, Vb, nAct);
  cvt_f32_bf16_kernel<<<nWgt / 1024, 256, 0, stream>>>(W_q, Wqb, nWgt);
  cvt_f32_bf16_kernel<<<nWgt / 1024, 256, 0, stream>>>(W_k, Wkb, nWgt);
  cvt_f32_bf16_kernel<<<nWgt / 1024, 256, 0, stream>>>(W_v, Wvb, nWgt);
  cvt_f32_bf16_kernel<<<nWgt / 1024, 256, 0, stream>>>(W_o, Wob, nWgt);

  dim3 ggrid(DM / 128, MROWS / 64);  // (8, 64)
  // q = Q @ Wq^T + bq  (bf16 row-major)
  gemm_nt_bf16_kernel<<<ggrid, 256, 0, stream>>>(Qb, Wqb, b_q, qproj, nullptr,
                                                 nullptr, MROWS, DM, DM);
  // k = K @ Wk^T + bk
  gemm_nt_bf16_kernel<<<ggrid, 256, 0, stream>>>(Kb, Wkb, b_k, kproj, nullptr,
                                                 nullptr, MROWS, DM, DM);
  // v = V @ Wv^T + bv  (stored transposed only: vT[DM][MROWS])
  gemm_nt_bf16_kernel<<<ggrid, 256, 0, stream>>>(Vb, Wvb, b_v, nullptr, nullptr,
                                                 vT, MROWS, DM, DM);

  // fused causal attention: scores, softmax, attn write, P@V
  attn_kernel<<<dim3(SEQ / 16, BATCH * HEADS), 256, 0, stream>>>(
      qproj, kproj, vT, attn_f, attnv);

  // out = attnv @ Wo^T + bo  (fp32 to d_out)
  gemm_nt_bf16_kernel<<<ggrid, 256, 0, stream>>>(attnv, Wob, b_o, nullptr, out_f,
                                                 nullptr, MROWS, DM, DM);
}
